// QLSTM_65481071398343
// MI455X (gfx1250) — matrix operand order repaired and checked by reading, not compiled, not run on hardware
//
#include <hip/hip_runtime.h>

// LSTM on MI455X (gfx1250, wave32, WMMA).
// Phase 1: W -> fp16, X -> fp16, init h/c state, zero grid barrier.
// Phase 2: Zx[S][4H][B] = X @ Wx.T + bias   (f16 WMMA, f32 acc, transposed f16 store)
// Phase 3: ONE persistent kernel loops over all 1024 timesteps:
//          - Wh slice resident in LDS for the whole kernel (staged once)
//          - per-step: wave-local h tile stage -> LDS; A & B frags via ds_load_b128
//          - fused sigmoid/tanh gating, c update, h write
//          - device-scope atomic grid barrier between steps (32 blocks, co-resident)

#define SEQ    1024
#define BATCH  256
#define IN_DIM 256
#define HID    256
#define KW     (IN_DIM + HID)   // 512
#define NG     4
#define GH     (NG * HID)       // 1024
#define LPAD   264              // LDS row stride in halves (bank-conflict padding)
#define NBLK   32               // persistent-kernel grid size

typedef __attribute__((ext_vector_type(16))) _Float16 v16h;
typedef __attribute__((ext_vector_type(8)))  float    v8f;

union Frag { unsigned int u[8]; uint4 q[2]; v16h h; };
union Pack4 { unsigned int u[4]; uint4 q; _Float16 hh[8]; };

__device__ __forceinline__ unsigned int pack2h(float a, float b) {
    union { _Float16 h[2]; unsigned int u; } p;
    p.h[0] = (_Float16)a; p.h[1] = (_Float16)b;
    return p.u;
}

__device__ __forceinline__ float sigmoidf_(float x) {
    return 1.0f / (1.0f + __expf(-x));
}

// ---- Phase 1a: W (fp32 [4][HID][KW]) -> fp16, gates ordered f,i,g,o ----
__global__ void convert_weights(const float* __restrict__ Wf,
                                const float* __restrict__ Wi,
                                const float* __restrict__ Wg,
                                const float* __restrict__ Wo,
                                _Float16* __restrict__ W16) {
    int idx = blockIdx.x * blockDim.x + threadIdx.x;   // 0 .. 4*HID*KW-1
    const int per = HID * KW;
    int g = idx / per;
    int r = idx - g * per;
    const float* W = (g == 0) ? Wf : (g == 1) ? Wi : (g == 2) ? Wg : Wo;
    W16[idx] = (_Float16)W[r];
}

// ---- Phase 1b: X (fp32) -> fp16, 4 elements per thread ----
__global__ void convert_x(const float4* __restrict__ X4,
                          uint2* __restrict__ X16) {
    int idx = blockIdx.x * blockDim.x + threadIdx.x;   // 0 .. S*B*I/4-1
    float4 v = X4[idx];
    uint2 r;
    r.x = pack2h(v.x, v.y);
    r.y = pack2h(v.z, v.w);
    X16[idx] = r;
}

// ---- Phase 1c: state init + barrier reset ----
__global__ void init_state(const float* __restrict__ h0,
                           const float* __restrict__ c0,
                           _Float16* __restrict__ h16,
                           float* __restrict__ cbuf,
                           unsigned* __restrict__ bar) {
    int idx = blockIdx.x * blockDim.x + threadIdx.x;   // 0 .. B*H-1
    h16[idx]  = (_Float16)h0[idx];
    cbuf[idx] = c0[idx];
    if (idx == 0) *bar = 0u;
}

// ---- Phase 2: Zx[S][4H][B] = X @ Wx.T + bias ----
// Block = 8 waves sharing one 16-row A tile (mtile = blockIdx>>1), ntile varies.
__global__ void __launch_bounds__(256)
lstm_xgemm(const _Float16* __restrict__ X16,   // [S*B, IN_DIM] fp16
           const _Float16* __restrict__ W16,   // [4][HID][KW] fp16
           const float* __restrict__ bf, const float* __restrict__ bi,
           const float* __restrict__ bg, const float* __restrict__ bo,
           _Float16* __restrict__ Zx)          // [S][4H][B] fp16
{
    __shared__ __align__(16) _Float16 xs[16 * LPAD];

    const int lane  = threadIdx.x & 31;
    const int wave  = threadIdx.x >> 5;
    const int wid   = blockIdx.x * 8 + wave;
    const int mtile = wid >> 4;                 // constant per block (= blockIdx>>1)
    const int ntile = wid & 15;
    const int l15   = lane & 15;
    const int hi16  = lane >> 4;
    const int ncol  = ntile * 16 + l15;         // B/C column for this lane
    const int kbA   = hi16 * 8;                 // A lane K base (ISA 16-bit A layout)
    const int kbB   = hi16 * 8;                 // B lane K base (ISA 16-bit B layout)

    // Cooperative stage of the 16x256 fp16 A tile (8 KB) into LDS, coalesced b128.
    const uint4* xsrc = (const uint4*)(X16 + (size_t)(blockIdx.x >> 1) * 16 * IN_DIM);
    #pragma unroll
    for (int i = threadIdx.x; i < 512; i += 256) {
        int row = i >> 5, c = i & 31;
        *(uint4*)(xs + row * LPAD + c * 8) = xsrc[row * 32 + c];
    }
    __syncthreads();

    const float* biasp[NG] = {bf, bi, bg, bo};
    v8f acc[NG];
    #pragma unroll
    for (int g = 0; g < NG; ++g) {
        float b = biasp[g][ncol];
        #pragma unroll
        for (int r = 0; r < 8; ++r) acc[g][r] = b;
    }

    const _Float16* xm = xs + l15 * LPAD;
    #pragma unroll
    for (int k0 = 0; k0 < IN_DIM; k0 += 32) {
        Frag a;
        a.q[0] = *(const uint4*)(xm + k0 + kbA);
        a.q[1] = *(const uint4*)(xm + k0 + kbA + 16);
        Frag b[NG];
        #pragma unroll
        for (int g = 0; g < NG; ++g) {
            const uint4* wv = (const uint4*)(W16 + ((size_t)(g * HID + ncol)) * KW + k0 + kbB);
            b[g].q[0] = wv[0];
            b[g].q[1] = wv[2];
        }
        #pragma unroll
        for (int g = 0; g < NG; ++g) {
            acc[g] = __builtin_amdgcn_wmma_f32_16x16x32_f16(
                false, a.h, false, b[g].h, (short)0, acc[g], false, false);
        }
    }

    // Store transposed: lane's 8 M-rows are contiguous -> one b128 per gate.
    const int gr = mtile * 16 + 8 * hi16;       // global row in S*B
    const int tt = gr >> 8;                     // timestep (tile never straddles t)
    const int bb = gr & 255;                    // batch index
    #pragma unroll
    for (int g = 0; g < NG; ++g) {
        Pack4 st;
        #pragma unroll
        for (int j = 0; j < 4; ++j)
            st.u[j] = pack2h(acc[g][2 * j], acc[g][2 * j + 1]);
        *(uint4*)(Zx + ((size_t)tt * GH + g * HID + ncol) * BATCH + bb) = st.q;
    }
}

// ---- Phase 3: persistent recurrence over all SEQ steps ----
// Block owns one hid-tile: th = blockIdx>>1; waves cover tb = 8*(blockIdx&1)+wave.
// Wh slice (32 KB) lives in LDS for the entire kernel.
__global__ void __launch_bounds__(256)
lstm_recurrence(const _Float16* __restrict__ W16,   // [4][HID][KW]; h-part at k>=IN_DIM
                const _Float16* __restrict__ Zx,    // [S][4H][B]
                _Float16* __restrict__ hA,          // [B,H] fp16 ping
                _Float16* __restrict__ hB,          // [B,H] fp16 pong
                float* __restrict__ cbuf,           // [B,H] fp32 state
                float* __restrict__ out,            // [S,B,H] ++ hx[B,H] ++ cx[B,H]
                unsigned* __restrict__ bar)         // grid barrier counter (starts 0)
{
    __shared__ __align__(16) _Float16 wsm[NG * 16 * LPAD];   // Wh slice: ~33 KB
    __shared__ __align__(16) _Float16 hs[8 * 16 * LPAD];     // per-wave h tiles: ~66 KB

    const int lane = threadIdx.x & 31;
    const int wave = threadIdx.x >> 5;
    const int th   = blockIdx.x >> 1;
    const int tb   = (blockIdx.x & 1) * 8 + wave;
    const int l15  = lane & 15;
    const int hi16 = lane >> 4;
    const int ncol = th * 16 + l15;
    const int kbA  = hi16 * 8;
    const int kbB  = hi16 * 8;
    const int rbase = tb * 16 + 8 * hi16;

    // Stage this block's Wh slice once: [4 gates][16 cols][K=256] halves.
    for (int i = threadIdx.x; i < 2048; i += 256) {
        int g = i >> 9, c = (i >> 5) & 15, q = i & 31;
        const uint4* src =
            (const uint4*)(W16 + ((size_t)(g * HID + th * 16 + c)) * KW + IN_DIM) + q;
        *(uint4*)(wsm + (g * 16 + c) * LPAD + q * 8) = *src;
    }
    __syncthreads();

    _Float16* hwave = hs + wave * 16 * LPAD;
    const _Float16* hm = hwave + l15 * LPAD;

    #pragma unroll 1
    for (int t = 0; t < SEQ; ++t) {
        const _Float16* hin = (t & 1) ? hB : hA;
        _Float16* hout      = (t & 1) ? hA : hB;

        // Wave-local stage of h_{t-1} rows [tb*16, tb*16+16): coalesced b128 per row.
        const uint4* hsrc = (const uint4*)(hin + (size_t)tb * 16 * HID);
        #pragma unroll
        for (int j = 0; j < 16; ++j)
            *(uint4*)(hwave + j * LPAD + lane * 8) = hsrc[j * 32 + lane];

        // C init from Zx_t (one b128 per gate) + prefetch next step's lines.
        v8f acc[NG];
        #pragma unroll
        for (int g = 0; g < NG; ++g) {
            const _Float16* zp = Zx + ((size_t)t * GH + g * HID + ncol) * BATCH + rbase;
            if (t + 1 < SEQ)
                __builtin_prefetch(zp + (size_t)GH * BATCH, 0, 0);
            Pack4 z;
            z.q = *(const uint4*)zp;
            #pragma unroll
            for (int r = 0; r < 8; ++r) acc[g][r] = (float)z.hh[r];
        }

        // acc += h_{t-1} @ Wh.T : A and B both from LDS (ds_load_b128).
        #pragma unroll
        for (int k0 = 0; k0 < HID; k0 += 32) {
            Frag a;
            a.q[0] = *(const uint4*)(hm + k0 + kbA);
            a.q[1] = *(const uint4*)(hm + k0 + kbA + 16);
            Frag b[NG];
            #pragma unroll
            for (int g = 0; g < NG; ++g) {
                const _Float16* wp = wsm + (g * 16 + l15) * LPAD + k0 + kbB;
                b[g].q[0] = *(const uint4*)wp;
                b[g].q[1] = *(const uint4*)(wp + 16);
            }
            #pragma unroll
            for (int g = 0; g < NG; ++g) {
                acc[g] = __builtin_amdgcn_wmma_f32_16x16x32_f16(
                    false, a.h, false, b[g].h, (short)0, acc[g], false, false);
            }
        }

        // Fused gating; each lane owns unique (b, h) cells -> race-free state update.
        #pragma unroll
        for (int r = 0; r < 8; ++r) {
            const size_t ci = (size_t)(rbase + r) * HID + ncol;
            float f  = sigmoidf_(acc[0][r]);
            float i  = sigmoidf_(acc[1][r]);
            float gg = tanhf(acc[2][r]);
            float o  = sigmoidf_(acc[3][r]);
            float cn = f * cbuf[ci] + i * gg;
            float hn = o * tanhf(cn);
            cbuf[ci] = cn;
            hout[ci] = (_Float16)hn;
            out[(size_t)t * BATCH * HID + ci] = hn;
            if (t == SEQ - 1) {
                out[(size_t)SEQ * BATCH * HID + ci] = hn;               // hx
                out[(size_t)SEQ * BATCH * HID + BATCH * HID + ci] = cn; // cx
            }
        }

        // Grid-wide barrier: release our stores, wait for all NBLK blocks.
        __syncthreads();
        if (threadIdx.x == 0) {
            __hip_atomic_fetch_add(bar, 1u, __ATOMIC_RELEASE, __HIP_MEMORY_SCOPE_AGENT);
            const unsigned target = (unsigned)(t + 1) * (unsigned)NBLK;
            while (__hip_atomic_load(bar, __ATOMIC_ACQUIRE, __HIP_MEMORY_SCOPE_AGENT)
                   < target)
                __builtin_amdgcn_s_sleep(1);
        }
        __syncthreads();
    }
}

extern "C" void kernel_launch(void* const* d_in, const int* in_sizes, int n_in,
                              void* d_out, int out_size, void* d_ws, size_t ws_size,
                              hipStream_t stream) {
    const float* X  = (const float*)d_in[0];   // [S,B,I]
    const float* h0 = (const float*)d_in[1];
    const float* c0 = (const float*)d_in[2];
    const float* Wf = (const float*)d_in[3];
    const float* bf = (const float*)d_in[4];
    const float* Wi = (const float*)d_in[5];
    const float* bi = (const float*)d_in[6];
    const float* Wg = (const float*)d_in[7];
    const float* bg = (const float*)d_in[8];
    const float* Wo = (const float*)d_in[9];
    const float* bo = (const float*)d_in[10];
    float* out = (float*)d_out;

    // Workspace layout (all offsets 16B-aligned by construction)
    char* ws = (char*)d_ws;
    size_t off = 0;
    _Float16* W16  = (_Float16*)(ws + off); off += (size_t)NG * HID * KW * 2;        // 1 MiB
    _Float16* h16a = (_Float16*)(ws + off); off += (size_t)BATCH * HID * 2;          // 128 KiB
    _Float16* h16b = (_Float16*)(ws + off); off += (size_t)BATCH * HID * 2;          // 128 KiB
    float*    cbuf = (float*)(ws + off);    off += (size_t)BATCH * HID * 4;          // 256 KiB
    unsigned* bar  = (unsigned*)(ws + off); off += 256;                              // barrier
    _Float16* X16  = (_Float16*)(ws + off); off += (size_t)SEQ * BATCH * IN_DIM * 2; // 128 MiB
    _Float16* Zx   = (_Float16*)(ws + off); // 512 MiB: [S][4H][B] fp16
    (void)ws_size; (void)in_sizes; (void)n_in; (void)out_size;

    convert_weights<<<(NG * HID * KW) / 256, 256, 0, stream>>>(Wf, Wi, Wg, Wo, W16);
    convert_x<<<(SEQ * BATCH * IN_DIM / 4) / 256, 256, 0, stream>>>(
        (const float4*)X, (uint2*)X16);
    init_state<<<(BATCH * HID) / 256, 256, 0, stream>>>(h0, c0, h16a, cbuf, bar);

    // Big parallel GEMM: (S*B/16) m-tiles x 16 n-tiles, 8 waves per block.
    const int xwaves = (SEQ * BATCH / 16) * (HID / 16);
    lstm_xgemm<<<xwaves / 8, 256, 0, stream>>>(X16, W16, bf, bi, bg, bo, Zx);

    // Single persistent kernel runs the whole recurrence (grid barrier per step).
    lstm_recurrence<<<NBLK, 256, 0, stream>>>(W16, Zx, h16a, h16b, cbuf, out, bar);
}
